// SGCNModel_70935679860746
// MI455X (gfx1250) — matrix operand order repaired and checked by reading, not compiled, not run on hardware
//
#include <hip/hip_runtime.h>
#include <hip/hip_bf16.h>
#include <math.h>
#include <stdint.h>

typedef __attribute__((ext_vector_type(16))) _Float16 v16h;
typedef __attribute__((ext_vector_type(8)))  float    v8f;
typedef __attribute__((ext_vector_type(4)))  int      v4i;

#define HDIM   128
#define H2DIM  64
#define LN_EPS 1e-5f
#define SAW    (HDIM + 4)   // padded LDS row stride (dwords): rows start on distinct banks

// Async global->LDS path if this toolchain declares the gfx1250 builtins.
#if defined(__HIP_DEVICE_COMPILE__) && __has_builtin(__builtin_amdgcn_global_load_async_to_lds_b128) && __has_builtin(__builtin_amdgcn_s_wait_asynccnt)
#define USE_ASYNC_LDS 1
#else
#define USE_ASYNC_LDS 0
#endif

#if USE_ASYNC_LDS
typedef __attribute__((address_space(1))) v4i gv4i;   // global (AS1)
typedef __attribute__((address_space(3))) v4i lv4i;   // LDS    (AS3)
#endif

__device__ __forceinline__ void stage16B(const float* g, float* l) {
#if USE_ASYNC_LDS
    // Generic pointer bits == AS1 address; LDS generic addr low 32 bits == LDS offset (ISA 10.2).
    gv4i* gp = (gv4i*)(unsigned long long)(uintptr_t)g;
    lv4i* lp = (lv4i*)(unsigned int)(uintptr_t)l;
    __builtin_amdgcn_global_load_async_to_lds_b128(gp, lp, 0, 0);
#else
    *(float4*)l = *(const float4*)g;
#endif
}
__device__ __forceinline__ void stage_wait() {
#if USE_ASYNC_LDS
    __builtin_amdgcn_s_wait_asynccnt(0);
#endif
}

// Hardware float atomic (global_atomic_add_f32), relaxed / device scope.
__device__ __forceinline__ void atomAddF(float* p, float v) {
    __hip_atomic_fetch_add(p, v, __ATOMIC_RELAXED, __HIP_MEMORY_SCOPE_AGENT);
}

// ---------------- WMMA fragment builders (wave32, V_WMMA_F32_16X16X32_F16) ----
// A (16x32, f16): lane L -> row m = L&15, half = L>>4.
//   slot s -> k = (s<8 ? s : s+8) + 8*half   (ISA 7.12.2 "16-bit A-Matrix 16x32")
__device__ __forceinline__ v16h load_a_frag(const float* row, int kb, int half) {
    v16h a;
#pragma unroll
    for (int s = 0; s < 16; ++s) {
        int k = kb + ((s < 8) ? s : (s + 8)) + 8 * half;
        a[s] = (_Float16)row[k];
    }
    return a;
}
// B (32x16, f16): lane L -> col n = L&15; slot s -> k = (s<8 ? s : s+8) + 8*half
__device__ __forceinline__ v16h load_b_frag(const float* wrow, int kb, int half) {
    v16h b;
#pragma unroll
    for (int s = 0; s < 16; ++s) b[s] = (_Float16)wrow[kb + ((s < 8) ? s : (s + 8)) + 8 * half];
    return b;
}

// ---------------- simple utility kernels ------------------------------------
__global__ void k_zero(float* __restrict__ p, int n) {
    int i = blockIdx.x * blockDim.x + threadIdx.x;
    if (i < n) p[i] = 0.0f;
}

__global__ void k_count(const int* __restrict__ dst, int E, float* __restrict__ cnt) {
    int e = blockIdx.x * blockDim.x + threadIdx.x;
    if (e < E) atomAddF(&cnt[dst[e]], 1.0f);
}

// Grid-stride, one wave per edge: scatter-add a 128-float row; prefetch the
// next edge's source row (global_prefetch_b8) while atomics are in flight.
__global__ void k_scatter128(const float* __restrict__ feat,
                             const int* __restrict__ src, const int* __restrict__ dst,
                             int E, float* __restrict__ out) {
    const int lane = threadIdx.x & 31;
    const int wid  = blockIdx.x * (blockDim.x >> 5) + (threadIdx.x >> 5);
    const int nw   = gridDim.x * (blockDim.x >> 5);
    for (int e = wid; e < E; e += nw) {
        const int en = e + nw;
        if (en < E && lane < 4)
            __builtin_prefetch(feat + (long long)src[en] * HDIM + lane * 32, 0, 0);
        const float* f = feat + (long long)src[e] * HDIM;
        float*       o = out  + (long long)dst[e] * HDIM;
#pragma unroll
        for (int i = 0; i < 4; ++i) {
            int c = lane + 32 * i;
            atomAddF(&o[c], f[c]);
        }
    }
}

// Layer-1 signed cross-wiring: cols 0..63 (xp) -> bufLo+offLo, cols 64..127 (xn) -> bufHi+offHi
__global__ void k_scatter_dual(const float* __restrict__ h0,
                               const int* __restrict__ src, const int* __restrict__ dst,
                               int E, float* __restrict__ bufLo, float* __restrict__ bufHi,
                               int offLo, int offHi) {
    const int lane = threadIdx.x & 31;
    const int wid  = blockIdx.x * (blockDim.x >> 5) + (threadIdx.x >> 5);
    const int nw   = gridDim.x * (blockDim.x >> 5);
    for (int e = wid; e < E; e += nw) {
        const int en = e + nw;
        if (en < E && lane < 4)
            __builtin_prefetch(h0 + (long long)src[en] * HDIM + lane * 32, 0, 0);
        const float* f = h0 + (long long)src[e] * HDIM;
        const long long d = (long long)dst[e] * HDIM;
#pragma unroll
        for (int i = 0; i < 2; ++i) {
            int c = lane + 32 * i;                 // 0..63  : xp
            atomAddF(&bufLo[d + offLo + c], f[c]);
        }
#pragma unroll
        for (int i = 2; i < 4; ++i) {
            int c = lane + 32 * i;                 // 64..127: xn
            atomAddF(&bufHi[d + offHi + (c - 64)], f[c]);
        }
    }
}

// sum -> mean:  cols 0..63 divide by cntLo, cols 64..127 by cntHi (clip >= 1)
__global__ void k_meandiv(float* __restrict__ buf, const float* __restrict__ cntLo,
                          const float* __restrict__ cntHi, int total) {
    int i = blockIdx.x * blockDim.x + threadIdx.x;
    if (i >= total) return;
    int n = i >> 7, c = i & 127;
    float cnt = (c < 64) ? cntLo[n] : cntHi[n];
    buf[i] = buf[i] / fmaxf(cnt, 1.0f);
}

// ---------------- fused signed-conv + ReLU + LayerNorm -----------------------
// out[:, 0:64]  = relu(aggP @ wPl^T + x[:,xoffP:] @ wPr^T + bP)
// out[:, 64:128]= relu(aggN @ wNl^T + x[:,xoffN:] @ wNr^T + bN)   then LN(128)
// 16 nodes per block; 8 waves = 8 output column tiles of 16.
// A-tiles (aggP/aggN each read by 4 waves, x by all 8) are staged once in LDS (async).
template <int K2>
__global__ __launch_bounds__(256) void k_sgconv_fused(
    const float* __restrict__ aggP, const float* __restrict__ aggN,
    const float* __restrict__ xb, int xoffP, int xoffN,
    const float* __restrict__ wPl, const float* __restrict__ wPr, const float* __restrict__ bP,
    const float* __restrict__ wNl, const float* __restrict__ wNr, const float* __restrict__ bN,
    const float* __restrict__ lng, const float* __restrict__ lnb,
    float* __restrict__ out, int N) {
    __shared__ float sA[3][16][SAW];      // aggP | aggN | x tiles (24.8 KB)
    __shared__ float hb[16][SAW];         // pre-LN activations
    const int tid   = threadIdx.x;
    const int wv    = tid >> 5;
    const int lane  = tid & 31;
    const int half  = lane >> 4;
    const int mr    = lane & 15;
    const int mbase = blockIdx.x * 16;

    // ---- stage the three 16x128 A-tiles into LDS ----
    {
        const float* srcs[3] = { aggP, aggN, xb };
#pragma unroll
        for (int t = 0; t < 3; ++t) {
            for (int i = tid; i < 512; i += 256) {         // 512 x 16B per tile
                int r = i >> 5, c4 = (i & 31) * 4;
                int rrow = mbase + r; if (rrow > N - 1) rrow = N - 1;
                stage16B(srcs[t] + (long long)rrow * HDIM + c4, &sA[t][r][c4]);
            }
        }
        stage_wait();
        __syncthreads();
    }

    const bool isPos  = (wv < 4);
    const int jbase   = (wv & 3) * 16;
    const int colbase = wv * 16;
    const float* wL   = isPos ? wPl : wNl;
    const float* wR   = isPos ? wPr : wNr;
    const float* bias = isPos ? bP  : bN;
    const int xoff    = isPos ? xoffP : xoffN;

    const float* arow  = &sA[isPos ? 0 : 1][mr][0];
    const float* xrow  = &sA[2][mr][xoff];
    const float* wLrow = wL + (jbase + mr) * HDIM;
    const float* wRrow = wR + (jbase + mr) * K2;

    v8f acc = {};
#pragma unroll
    for (int kb = 0; kb < HDIM; kb += 32) {
        v16h a = load_a_frag(arow, kb, half);
        v16h b = load_b_frag(wLrow, kb, half);
        acc = __builtin_amdgcn_wmma_f32_16x16x32_f16(false, a, false, b, (short)0, acc, false, false);
    }
#pragma unroll
    for (int kb = 0; kb < K2; kb += 32) {
        v16h a = load_a_frag(xrow, kb, half);
        v16h b = load_b_frag(wRrow, kb, half);
        acc = __builtin_amdgcn_wmma_f32_16x16x32_f16(false, a, false, b, (short)0, acc, false, false);
    }
    const float bj = bias[jbase + mr];
#pragma unroll
    for (int r = 0; r < 8; ++r) {
        int m = r + 8 * half;                     // C/D layout: m = r + 8*half, n = lane&15
        hb[m][colbase + mr] = fmaxf(acc[r] + bj, 0.0f);
    }
    __syncthreads();

    // LayerNorm over 128: 16 threads per row, 8 cols each, shfl_xor reduce within 16 lanes.
    const int rr = tid >> 4;
    const int cs = (tid & 15) * 8;
    float vals[8];
    float s = 0.0f;
#pragma unroll
    for (int i = 0; i < 8; ++i) { vals[i] = hb[rr][cs + i]; s += vals[i]; }
    s += __shfl_xor(s, 1, 16); s += __shfl_xor(s, 2, 16);
    s += __shfl_xor(s, 4, 16); s += __shfl_xor(s, 8, 16);
    const float mu = s * (1.0f / 128.0f);
    float q = 0.0f;
#pragma unroll
    for (int i = 0; i < 8; ++i) { float d = vals[i] - mu; q += d * d; }
    q += __shfl_xor(q, 1, 16); q += __shfl_xor(q, 2, 16);
    q += __shfl_xor(q, 4, 16); q += __shfl_xor(q, 8, 16);
    const float rstd = rsqrtf(q * (1.0f / 128.0f) + LN_EPS);
    const int grow = mbase + rr;
    if (grow < N) {
        float* orow = out + (long long)grow * HDIM;
#pragma unroll
        for (int i = 0; i < 8; ++i) {
            int c = cs + i;
            orow[c] = (vals[i] - mu) * rstd * lng[c] + lnb[c];
        }
    }
}

// ---------------- fused MLP: gelu(h@pw1^T+pb1)@pw2^T+pb2, gated mix with x0 --
__global__ __launch_bounds__(256) void k_mlp_fused(
    const float* __restrict__ h1,
    const float* __restrict__ pw1, const float* __restrict__ pb1,
    const float* __restrict__ pw2, const float* __restrict__ pb2,
    const float* __restrict__ gate, const float* __restrict__ x0,
    float* __restrict__ out, int N) {
    __shared__ float sH[16][SAW];         // staged h1 tile (read by all 8 waves)
    __shared__ float tb[16][SAW];         // post-GELU intermediate
    const int tid   = threadIdx.x;
    const int wv    = tid >> 5;
    const int lane  = tid & 31;
    const int half  = lane >> 4;
    const int mr    = lane & 15;
    const int mbase = blockIdx.x * 16;
    const int jbase = wv * 16;

    for (int i = tid; i < 512; i += 256) {
        int r = i >> 5, c4 = (i & 31) * 4;
        int rrow = mbase + r; if (rrow > N - 1) rrow = N - 1;
        stage16B(h1 + (long long)rrow * HDIM + c4, &sH[r][c4]);
    }
    stage_wait();
    __syncthreads();

    const float* arow  = &sH[mr][0];
    const float* w1row = pw1 + (jbase + mr) * HDIM;
    v8f acc = {};
#pragma unroll
    for (int kb = 0; kb < HDIM; kb += 32) {
        v16h a = load_a_frag(arow, kb, half);
        v16h b = load_b_frag(w1row, kb, half);
        acc = __builtin_amdgcn_wmma_f32_16x16x32_f16(false, a, false, b, (short)0, acc, false, false);
    }
    const float b1 = pb1[jbase + mr];
#pragma unroll
    for (int r = 0; r < 8; ++r) {
        int m = r + 8 * half;
        float v = acc[r] + b1;
        tb[m][jbase + mr] = 0.5f * v * (1.0f + erff(v * 0.70710678118654752440f)); // exact GELU
    }
    __syncthreads();

    const float* trow  = &tb[mr][0];               // A for 2nd GEMM straight from LDS
    const float* w2row = pw2 + (jbase + mr) * HDIM;
    v8f acc2 = {};
#pragma unroll
    for (int kb = 0; kb < HDIM; kb += 32) {
        v16h a = load_a_frag(trow, kb, half);
        v16h b = load_b_frag(w2row, kb, half);
        acc2 = __builtin_amdgcn_wmma_f32_16x16x32_f16(false, a, false, b, (short)0, acc2, false, false);
    }
    const float gt = gate[0];
    const float b2 = pb2[jbase + mr];
#pragma unroll
    for (int r = 0; r < 8; ++r) {
        int g = mbase + r + 8 * half;
        if (g < N) {
            long long idx = (long long)g * HDIM + (jbase + mr);
            out[idx] = gt * (acc2[r] + b2) + (1.0f - gt) * x0[idx];
        }
    }
}

// ---------------- host launcher ----------------------------------------------
extern "C" void kernel_launch(void* const* d_in, const int* in_sizes, int n_in,
                              void* d_out, int out_size, void* d_ws, size_t ws_size,
                              hipStream_t stream) {
    const float* x       = (const float*)d_in[0];
    const int*   pei     = (const int*)d_in[1];
    const int*   nei     = (const int*)d_in[2];
    const float* c0_pl_w = (const float*)d_in[3];
    const float* c0_pr_w = (const float*)d_in[4];
    const float* c0_pr_b = (const float*)d_in[5];
    const float* c0_nl_w = (const float*)d_in[6];
    const float* c0_nr_w = (const float*)d_in[7];
    const float* c0_nr_b = (const float*)d_in[8];
    const float* c1_pl_w = (const float*)d_in[9];
    const float* c1_pr_w = (const float*)d_in[10];
    const float* c1_pr_b = (const float*)d_in[11];
    const float* c1_nl_w = (const float*)d_in[12];
    const float* c1_nr_w = (const float*)d_in[13];
    const float* c1_nr_b = (const float*)d_in[14];
    const float* ln0_g   = (const float*)d_in[15];
    const float* ln0_b   = (const float*)d_in[16];
    const float* ln1_g   = (const float*)d_in[17];
    const float* ln1_b   = (const float*)d_in[18];
    const float* pw1     = (const float*)d_in[19];
    const float* pb1     = (const float*)d_in[20];
    const float* pw2     = (const float*)d_in[21];
    const float* pb2     = (const float*)d_in[22];
    const float* gate    = (const float*)d_in[23];
    float* out = (float*)d_out;

    const int N = in_sizes[0] / HDIM;
    const int E = in_sizes[1] / 2;
    const long long NH = (long long)N * HDIM;

    float* ws   = (float*)d_ws;
    float* sumP = ws;                 // N x 128
    float* sumN = ws + NH;            // N x 128
    float* h0   = ws + 2 * NH;        // N x 128
    float* h1   = ws + 3 * NH;        // N x 128
    float* cntP = ws + 4 * NH;        // N
    float* cntN = cntP + N;           // N

    const int* psrc = pei;  const int* pdst = pei + E;
    const int* nsrc = nei;  const int* ndst = nei + E;

    const int ztot = (int)(2 * NH);
    const int tot  = (int)NH;
    const int edgeBlocks = 4096;          // grid-stride: 32768 waves over E edges
    const int tiles = (N + 15) / 16;

    // ---- Layer 0 aggregation ----
    k_zero<<<(ztot + 255) / 256, 256, 0, stream>>>(sumP, ztot);
    k_zero<<<(2 * N + 255) / 256, 256, 0, stream>>>(cntP, 2 * N);
    k_count<<<(E + 255) / 256, 256, 0, stream>>>(pdst, E, cntP);
    k_count<<<(E + 255) / 256, 256, 0, stream>>>(ndst, E, cntN);
    k_scatter128<<<edgeBlocks, 256, 0, stream>>>(x, psrc, pdst, E, sumP);
    k_scatter128<<<edgeBlocks, 256, 0, stream>>>(x, nsrc, ndst, E, sumN);
    k_meandiv<<<(tot + 255) / 256, 256, 0, stream>>>(sumP, cntP, cntP, tot);
    k_meandiv<<<(tot + 255) / 256, 256, 0, stream>>>(sumN, cntN, cntN, tot);

    // ---- Layer 0 conv + ReLU + LN -> h0 ----
    k_sgconv_fused<HDIM><<<tiles, 256, 0, stream>>>(
        sumP, sumN, x, 0, 0,
        c0_pl_w, c0_pr_w, c0_pr_b, c0_nl_w, c0_nr_w, c0_nr_b,
        ln0_g, ln0_b, h0, N);

    // ---- Layer 1 aggregation (signed cross-wiring) ----
    k_zero<<<(ztot + 255) / 256, 256, 0, stream>>>(sumP, ztot);
    // pos edges: xp -> pos_msg[:, :64], xn -> neg_msg[:, :64]
    k_scatter_dual<<<edgeBlocks, 256, 0, stream>>>(h0, psrc, pdst, E, sumP, sumN, 0, 0);
    // neg edges: xp -> neg_msg[:, 64:], xn -> pos_msg[:, 64:]
    k_scatter_dual<<<edgeBlocks, 256, 0, stream>>>(h0, nsrc, ndst, E, sumN, sumP, 64, 64);
    k_meandiv<<<(tot + 255) / 256, 256, 0, stream>>>(sumP, cntP, cntN, tot);
    k_meandiv<<<(tot + 255) / 256, 256, 0, stream>>>(sumN, cntP, cntN, tot);

    // ---- Layer 1 conv + ReLU + LN -> h1 (x-part reads xp/xn slices of h0) ----
    k_sgconv_fused<H2DIM><<<tiles, 256, 0, stream>>>(
        sumP, sumN, h0, 0, H2DIM,
        c1_pl_w, c1_pr_w, c1_pr_b, c1_nl_w, c1_nr_w, c1_nr_b,
        ln1_g, ln1_b, h1, N);

    // ---- Fused MLP + gated residual -> out ----
    k_mlp_fused<<<tiles, 256, 0, stream>>>(h1, pw1, pb1, pw2, pb2, gate, x, out, N);
}
